// AttnBlock_7756710936949
// MI455X (gfx1250) — matrix operand order repaired and checked by reading, not compiled, not run on hardware
//
#include <hip/hip_runtime.h>

// ---------------------------------------------------------------------------
// AttnBlock for MI455X (gfx1250, wave32, WMMA).
// b=4, c=512, h=w=64, n=4096.  All GEMM-like math through
// v_wmma_f32_16x16x32_bf16 (bf16 operands, f32 accumulation).
// Attention is flash-style (online softmax), scores never hit memory.
// Q tile staged to LDS with gfx1250 async-to-LDS loads (ASYNCcnt).
// ---------------------------------------------------------------------------

typedef __attribute__((ext_vector_type(16))) __bf16 v16bf;
typedef __attribute__((ext_vector_type(8)))  float  v8f;

union Frag {
  v16bf v;
  int4  q[2];
  unsigned short u[16];
};

__device__ __forceinline__ unsigned short f2bf(float f) {
  unsigned int u = __float_as_uint(f);
  u += 0x7FFFu + ((u >> 16) & 1u);   // round-to-nearest-even
  return (unsigned short)(u >> 16);
}
__device__ __forceinline__ float bf2f(unsigned short h) {
  return __uint_as_float(((unsigned int)h) << 16);
}

__device__ __forceinline__ v8f wmma_bf16(const Frag& a, const Frag& b, v8f c) {
  return __builtin_amdgcn_wmma_f32_16x16x32_bf16(
      /*neg_a=*/false, a.v, /*neg_b=*/false, b.v,
      /*c_mod=*/(short)0, c, /*reuse_a=*/false, /*reuse_b=*/false);
}

static constexpr int Cc = 512;
static constexpr int Nn = 4096;   // 64*64 spatial positions

// ---------------------------------------------------------------------------
// f32 -> bf16 conversion (weights)
// ---------------------------------------------------------------------------
__global__ void f32_to_bf16_kernel(const float* __restrict__ src,
                                   unsigned short* __restrict__ dst, int n) {
  int i = blockIdx.x * 256 + threadIdx.x;
  if (i < n) dst[i] = f2bf(src[i]);
}

// ---------------------------------------------------------------------------
// GroupNorm (32 groups).  One block per (batch, group): 16 channels x 4096.
// Writes hnT[b][pos][ch] in bf16 -> ready-made [N,K] row-major WMMA-B operand.
// ---------------------------------------------------------------------------
__global__ __launch_bounds__(256)
void groupnorm_kernel(const float* __restrict__ x,
                      const float* __restrict__ gamma,
                      const float* __restrict__ beta,
                      unsigned short* __restrict__ hnT) {
  const int bg = blockIdx.x;            // 0..127
  const int b  = bg >> 5;
  const int g  = bg & 31;
  const int c0 = g * 16;
  const float* xg = x + ((size_t)b * Cc + c0) * Nn;

  float s = 0.f, ss = 0.f;
  for (int i = threadIdx.x; i < 16 * Nn; i += 256) {
    float v = xg[i];
    s += v;  ss += v * v;
  }
  // wave32 reduce then cross-wave via LDS
  #pragma unroll
  for (int off = 16; off > 0; off >>= 1) {
    s  += __shfl_down(s,  off, 32);
    ss += __shfl_down(ss, off, 32);
  }
  __shared__ float red0[8], red1[8];
  const int wave = threadIdx.x >> 5, lane = threadIdx.x & 31;
  if (lane == 0) { red0[wave] = s; red1[wave] = ss; }
  __syncthreads();
  if (threadIdx.x == 0) {
    float ts = 0.f, tss = 0.f;
    #pragma unroll
    for (int w = 0; w < 8; ++w) { ts += red0[w]; tss += red1[w]; }
    red0[0] = ts; red1[0] = tss;
  }
  __syncthreads();
  const float inv_n = 1.0f / (16.0f * (float)Nn);
  const float mean  = red0[0] * inv_n;
  const float var   = red1[0] * inv_n - mean * mean;
  const float rstd  = rsqrtf(var + 1e-6f);

  for (int i = threadIdx.x; i < 16 * Nn; i += 256) {
    const int ch  = i >> 12;       // /4096
    const int pos = i & (Nn - 1);
    const int c   = c0 + ch;
    float v = (xg[i] - mean) * rstd * gamma[c] + beta[c];
    hnT[((size_t)b * Nn + pos) * Cc + c] = f2bf(v);
  }
}

// ---------------------------------------------------------------------------
// WMMA GEMM:  D[m,n] = sum_k A[m,k] * B[b][n,k]  + bias[m]   (M=K=512, N=4096)
//   A: weights [512,512] bf16 row-major (K contiguous)  -> A-fragments
//   B: activations [b][4096,512] bf16 row-major          -> B-fragments
// Block: 8 waves (2x4), each wave computes a 32x32 tile (2x2 WMMA tiles).
// mode 0: store bf16 transposed outT[b][n][m]   (qT / kT)
// mode 1: store bf16 direct     out [b][m][n]   (V)
// mode 2: store f32 out[b][m][n] = acc + bias + resid (proj + residual)
// ---------------------------------------------------------------------------
__global__ __launch_bounds__(256)
void gemm_bf16_kernel(const unsigned short* __restrict__ A,
                      const unsigned short* __restrict__ Bm,
                      const float* __restrict__ bias,
                      void* __restrict__ out,
                      const float* __restrict__ resid,
                      int mode) {
  const int lane = threadIdx.x & 31;
  const int wv   = threadIdx.x >> 5;
  const int wm   = wv & 1;
  const int wn   = wv >> 1;
  const int m0   = blockIdx.x * 64  + wm * 32;
  const int n0   = blockIdx.y * 128 + wn * 32;
  const int bb   = blockIdx.z;

  const unsigned short* Bb = Bm + (size_t)bb * Nn * Cc;
  const int ra = lane & 15;
  const int ka = (lane >> 4) * 8;    // A-frag K base (ISA 7.12.2, 16-bit A)
  const int kb = (lane >> 4) * 8;    // B-frag K base (same K order as A)

  const v8f zero = {0.f,0.f,0.f,0.f,0.f,0.f,0.f,0.f};
  v8f acc00 = zero, acc01 = zero, acc10 = zero, acc11 = zero;

  const unsigned short* a0p = A  + (size_t)(m0 + ra) * Cc;
  const unsigned short* a1p = a0p + 16 * Cc;
  const unsigned short* b0p = Bb + (size_t)(n0 + ra) * Cc;
  const unsigned short* b1p = b0p + 16 * Cc;

  for (int k = 0; k < Cc; k += 32) {
    Frag a0, a1, b0, b1;
    a0.q[0] = *(const int4*)(a0p + k + ka);
    a0.q[1] = *(const int4*)(a0p + k + ka + 16);
    a1.q[0] = *(const int4*)(a1p + k + ka);
    a1.q[1] = *(const int4*)(a1p + k + ka + 16);
    b0.q[0] = *(const int4*)(b0p + k + kb);
    b0.q[1] = *(const int4*)(b0p + k + kb + 16);
    b1.q[0] = *(const int4*)(b1p + k + kb);
    b1.q[1] = *(const int4*)(b1p + k + kb + 16);
    acc00 = wmma_bf16(a0, b0, acc00);
    acc01 = wmma_bf16(a0, b1, acc01);
    acc10 = wmma_bf16(a1, b0, acc10);
    acc11 = wmma_bf16(a1, b1, acc11);
  }

  // C/D layout: lanes 0-15 -> M=r, N=lane; lanes 16-31 -> M=8+r, N=lane-16
  const int nn = lane & 15;
  const int mh = (lane >> 4) * 8;
  v8f accs[2][2] = {{acc00, acc01}, {acc10, acc11}};

  #pragma unroll
  for (int tm = 0; tm < 2; ++tm) {
    #pragma unroll
    for (int tn = 0; tn < 2; ++tn) {
      #pragma unroll
      for (int r = 0; r < 8; ++r) {
        const int m = m0 + tm * 16 + mh + r;
        const int n = n0 + tn * 16 + nn;
        const float v = accs[tm][tn][r] + bias[m];
        if (mode == 0) {
          ((unsigned short*)out)[((size_t)bb * Nn + n) * Cc + m] = f2bf(v);
        } else if (mode == 1) {
          ((unsigned short*)out)[((size_t)bb * Cc + m) * Nn + n] = f2bf(v);
        } else {
          const size_t idx = ((size_t)bb * Cc + m) * Nn + n;
          ((float*)out)[idx] = v + resid[idx];
        }
      }
    }
  }
}

// ---------------------------------------------------------------------------
// Flash attention.  Grid: (n/16 query tiles, 2 channel halves, b).
// One wave per block: 16 queries x 256 output channels, online softmax.
// qT,kT: [b][n][c] bf16;  V: [b][c][n] bf16;  OT: [b][n][c] bf16.
// ---------------------------------------------------------------------------
__global__ __launch_bounds__(32)
void attn_fa_kernel(const unsigned short* __restrict__ qT,
                    const unsigned short* __restrict__ kT,
                    const unsigned short* __restrict__ Vm,
                    unsigned short* __restrict__ OT) {
  __shared__ __align__(16) unsigned short qs[16 * Cc];   // 16 KB query tile
  __shared__ __align__(16) unsigned short Ps[16 * 32];   // 1 KB  P tile

  const int lane    = threadIdx.x;
  const int i0      = blockIdx.x * 16;
  const int ch_base = blockIdx.y * 256;
  const int bb      = blockIdx.z;

  // ---- stage Q tile (16 rows x 512 bf16 = 16 KB) into LDS via the gfx1250
  //      async data-mover path: global_load_async_to_lds_b128 (ASYNCcnt),
  //      no VGPR round-trip.  LDS offset = low 32 bits of generic pointer.
  {
    const unsigned short* qrow = qT + ((size_t)bb * Nn + i0) * Cc;
    unsigned lds_base = (unsigned)(uintptr_t)(void*)qs;
    #pragma unroll 4
    for (int t = lane; t < (16 * Cc) / 8; t += 32) {
      unsigned dst = lds_base + (unsigned)t * 16u;
      const unsigned short* src = qrow + t * 8;
      asm volatile("global_load_async_to_lds_b128 %0, %1, off"
                   :: "v"(dst), "v"(src) : "memory");
    }
    asm volatile("s_wait_asynccnt 0x0" ::: "memory");
  }
  __syncthreads();

  const int ra = lane & 15;
  const int ka = (lane >> 4) * 8;
  const int kb = (lane >> 4) * 8;
  const v8f zero = {0.f,0.f,0.f,0.f,0.f,0.f,0.f,0.f};

  float mrow[8], lrow[8];
  v8f acc[16];
  #pragma unroll
  for (int r = 0; r < 8; ++r) { mrow[r] = -1e30f; lrow[r] = 0.f; }
  #pragma unroll
  for (int t = 0; t < 16; ++t) acc[t] = zero;

  const float scale = 0.044194173824159216f;   // 512^-0.5
  const unsigned short* vb = Vm + (size_t)bb * Cc * Nn;

  for (int j0 = 0; j0 < Nn; j0 += 32) {
    const unsigned short* krow = kT + ((size_t)bb * Nn + j0) * Cc;
    // prefetch next key tile (gfx1250 global_prefetch_b8)
    if (j0 + 32 < Nn) __builtin_prefetch(krow + 32 * Cc + lane * 32, 0, 0);

    // ---- scores: S[16 q x 32 k] = Q(16x512) * K^T, 32 WMMAs ----
    v8f s0 = zero, s1 = zero;
    #pragma unroll 2
    for (int k = 0; k < Cc; k += 32) {
      Frag a, b0, b1;
      const unsigned short* qp = qs + ra * Cc + k;
      a.q[0] = *(const int4*)(qp + ka);
      a.q[1] = *(const int4*)(qp + ka + 16);
      const unsigned short* kp = krow + (size_t)ra * Cc + k;
      b0.q[0] = *(const int4*)(kp + kb);
      b0.q[1] = *(const int4*)(kp + kb + 16);
      const unsigned short* kp1 = kp + 16 * Cc;
      b1.q[0] = *(const int4*)(kp1 + kb);
      b1.q[1] = *(const int4*)(kp1 + kb + 16);
      s0 = wmma_bf16(a, b0, s0);
      s1 = wmma_bf16(a, b1, s1);
    }

    // ---- online softmax on D-layout tiles (rows across 16-lane halves) ----
    float corr[8];
    #pragma unroll
    for (int r = 0; r < 8; ++r) {
      const float a0 = s0[r] * scale;
      const float a1 = s1[r] * scale;
      float v = fmaxf(a0, a1);
      v = fmaxf(v, __shfl_xor(v, 1, 32));
      v = fmaxf(v, __shfl_xor(v, 2, 32));
      v = fmaxf(v, __shfl_xor(v, 4, 32));
      v = fmaxf(v, __shfl_xor(v, 8, 32));
      const float mn = fmaxf(mrow[r], v);
      const float cr = __expf(mrow[r] - mn);
      mrow[r] = mn;
      const float p0 = __expf(a0 - mn);
      const float p1 = __expf(a1 - mn);
      float rs = p0 + p1;
      rs += __shfl_xor(rs, 1, 32);
      rs += __shfl_xor(rs, 2, 32);
      rs += __shfl_xor(rs, 4, 32);
      rs += __shfl_xor(rs, 8, 32);
      lrow[r] = lrow[r] * cr + rs;
      corr[r] = cr;
      s0[r] = p0;
      s1[r] = p1;
    }

    // ---- D-layout -> A-layout via LDS (P as bf16 [16 x 32]) ----
    const int col = lane & 15;
    const int rb  = (lane >> 4) * 8;
    #pragma unroll
    for (int r = 0; r < 8; ++r) {
      Ps[(rb + r) * 32 + col]      = f2bf(s0[r]);
      Ps[(rb + r) * 32 + col + 16] = f2bf(s1[r]);
    }
    __syncthreads();

    Frag pa;
    const unsigned short* pp = Ps + ra * 32;
    pa.q[0] = *(const int4*)(pp + ka);
    pa.q[1] = *(const int4*)(pp + ka + 16);

    // ---- rescale accumulators + PV: O += P(16x32) * V^T(32x256), 16 WMMAs
    #pragma unroll
    for (int t = 0; t < 16; ++t) {
      #pragma unroll
      for (int r = 0; r < 8; ++r) acc[t][r] *= corr[r];
      Frag bf_;
      const unsigned short* vp = vb + (size_t)(ch_base + t * 16 + ra) * Nn + j0;
      bf_.q[0] = *(const int4*)(vp + kb);
      bf_.q[1] = *(const int4*)(vp + kb + 16);
      acc[t] = wmma_bf16(pa, bf_, acc[t]);
    }
  }

  // ---- normalize and store O^T[b][i][ch] (B-operand layout for proj) ----
  #pragma unroll
  for (int r = 0; r < 8; ++r) lrow[r] = 1.0f / lrow[r];
  const int colo = lane & 15;
  const int mh   = (lane >> 4) * 8;
  #pragma unroll
  for (int t = 0; t < 16; ++t) {
    #pragma unroll
    for (int r = 0; r < 8; ++r) {
      const int row = i0 + mh + r;
      const int ch  = ch_base + t * 16 + colo;
      OT[((size_t)bb * Nn + row) * Cc + ch] = f2bf(acc[t][r] * lrow[r]);
    }
  }
}

// ---------------------------------------------------------------------------
// Launch
// ---------------------------------------------------------------------------
extern "C" void kernel_launch(void* const* d_in, const int* in_sizes, int n_in,
                              void* d_out, int out_size, void* d_ws, size_t ws_size,
                              hipStream_t stream) {
  (void)in_sizes; (void)n_in; (void)out_size; (void)ws_size;

  const float* x     = (const float*)d_in[0];
  const float* gamma = (const float*)d_in[1];
  const float* beta  = (const float*)d_in[2];
  const float* wq    = (const float*)d_in[3];
  const float* bq    = (const float*)d_in[4];
  const float* wk    = (const float*)d_in[5];
  const float* bk    = (const float*)d_in[6];
  const float* wv    = (const float*)d_in[7];
  const float* bv    = (const float*)d_in[8];
  const float* wp    = (const float*)d_in[9];
  const float* bp    = (const float*)d_in[10];

  char* ws = (char*)d_ws;
  const size_t WSZ = (size_t)Cc * Cc * 2;          // one bf16 weight: 512 KB
  const size_t ASZ = (size_t)4 * Nn * Cc * 2;      // one bf16 activation: ~16.8 MB
  unsigned short* wq_b = (unsigned short*)(ws + 0 * WSZ);
  unsigned short* wk_b = (unsigned short*)(ws + 1 * WSZ);
  unsigned short* wv_b = (unsigned short*)(ws + 2 * WSZ);
  unsigned short* wp_b = (unsigned short*)(ws + 3 * WSZ);
  unsigned short* hnT  = (unsigned short*)(ws + 4 * WSZ + 0 * ASZ);
  unsigned short* qT   = (unsigned short*)(ws + 4 * WSZ + 1 * ASZ);
  unsigned short* kT   = (unsigned short*)(ws + 4 * WSZ + 2 * ASZ);
  unsigned short* Vm   = (unsigned short*)(ws + 4 * WSZ + 3 * ASZ);
  unsigned short* OT   = (unsigned short*)(ws + 4 * WSZ + 4 * ASZ);

  const int wn = Cc * Cc;
  f32_to_bf16_kernel<<<(wn + 255) / 256, 256, 0, stream>>>(wq, wq_b, wn);
  f32_to_bf16_kernel<<<(wn + 255) / 256, 256, 0, stream>>>(wk, wk_b, wn);
  f32_to_bf16_kernel<<<(wn + 255) / 256, 256, 0, stream>>>(wv, wv_b, wn);
  f32_to_bf16_kernel<<<(wn + 255) / 256, 256, 0, stream>>>(wp, wp_b, wn);

  groupnorm_kernel<<<128, 256, 0, stream>>>(x, gamma, beta, hnT);

  dim3 gg(8, 32, 4);   // M/64, N/128, batch
  gemm_bf16_kernel<<<gg, 256, 0, stream>>>(wq_b, hnT, bq, (void*)qT, nullptr, 0);
  gemm_bf16_kernel<<<gg, 256, 0, stream>>>(wk_b, hnT, bk, (void*)kT, nullptr, 0);
  gemm_bf16_kernel<<<gg, 256, 0, stream>>>(wv_b, hnT, bv, (void*)Vm, nullptr, 1);

  dim3 ga(Nn / 16, 2, 4);   // query tiles, channel halves, batch
  attn_fa_kernel<<<ga, 32, 0, stream>>>(qT, kT, Vm, OT);

  gemm_bf16_kernel<<<gg, 256, 0, stream>>>(wp_b, OT, bp, d_out, x, 2);
}
